// Agent_81801947119732
// MI455X (gfx1250) — matrix operand order repaired and checked by reading, not compiled, not run on hardware
//
#include <hip/hip_runtime.h>
#include <hip/hip_bf16.h>
#include <math.h>

typedef __attribute__((ext_vector_type(16))) __bf16 v16bf;
typedef __attribute__((ext_vector_type(8)))  __bf16 v8bf;
typedef __attribute__((ext_vector_type(8)))  float  v8f;

static constexpr int T_  = 512;
static constexpr int B_  = 64;
static constexpr int D_  = 64;
static constexpr int H_  = 512;
static constexpr int R_  = 512;
static constexpr int A_  = 18;
static constexpr int TB_ = T_ * B_;   // 32768

// ---------------------------------------------------------------------------
// helpers
// ---------------------------------------------------------------------------
__device__ __forceinline__ __bf16 f2bf(float v) { return (__bf16)v; }

__device__ __forceinline__ v8f wmma_bf16(v16bf a, v16bf b, v8f c) {
  // D = A(16x32 bf16) * B(32x16 bf16) + C(16x16 f32)
  return __builtin_amdgcn_wmma_f32_16x16x32_bf16(
      /*neg_a=*/false, a, /*neg_b=*/false, b,
      /*c_mod=*/(short)0, c, /*reuse_a=*/false, /*reuse_b=*/false);
}

// Assemble an A fragment from a per-lane base pointer (two 16B loads).
// Caller bakes in: p = A + (m0 + lane%16)*lda + (lane<16 ? 0 : 8), then + k0.
__device__ __forceinline__ v16bf make_a(const __bf16* p) {
  v8bf lo = *(const v8bf*)(p);        // K = kb + 0..7
  v8bf hi = *(const v8bf*)(p + 16);   // K = kb + 16..23
  v16bf a;
#pragma unroll
  for (int j = 0; j < 8; ++j) { a[j] = lo[j]; a[j + 8] = hi[j]; }
  return a;
}

__device__ __forceinline__ float sigmoidf_(float x) { return 1.f / (1.f + expf(-x)); }

// ---------------------------------------------------------------------------
// converts / packers
// ---------------------------------------------------------------------------
__global__ void cvt_bf16_kernel(const float* __restrict__ in,
                                __bf16* __restrict__ out, int n) {
  int i = blockIdx.x * blockDim.x + threadIdx.x;
  if (i < n) out[i] = f2bf(in[i]);
}

// convert with per-row done-mask (rows of length rowlen)
__global__ void cvt_bf16_mask_kernel(const float* __restrict__ in,
                                     __bf16* __restrict__ out,
                                     const int* __restrict__ done,
                                     int n, int rowlen) {
  int i = blockIdx.x * blockDim.x + threadIdx.x;
  if (i < n) {
    int m = i / rowlen;
    out[i] = done[m] ? (__bf16)0.f : f2bf(in[i]);
  }
}

// Convert + repack f32 weight [K,N] into WMMA-B fragment-native layout:
// out[((k0/32)*ntiles + n0/16)*512 + lane*16 + j] = B[k, n]
//   k = k0 + (j<8 ? 0 : 16) + (lane<16 ? 0 : 8) + (j & 7) ,  n = n0 + (lane & 15)
__global__ void pack_b_kernel(const float* __restrict__ B,
                              __bf16* __restrict__ out, int K, int N) {
  int idx = blockIdx.x * blockDim.x + threadIdx.x;
  if (idx >= K * N) return;
  int j    = idx & 15;
  int lane = (idx >> 4) & 31;
  int tile = idx >> 9;
  int ntn  = N >> 4;
  int ks   = tile / ntn;
  int nt   = tile - ks * ntn;
  int k    = ks * 32 + ((j >> 3) << 4) + ((lane >> 4) << 3) + (j & 7);
  int n    = nt * 16 + (lane & 15);
  out[idx] = f2bf(B[(size_t)k * N + n]);
}

// actor_w [R,A] -> [A,R]
__global__ void transpose_aw_kernel(const float* __restrict__ w,
                                    float* __restrict__ wt) {
  int i = blockIdx.x * blockDim.x + threadIdx.x;
  if (i < R_ * A_) {
    int k = i / A_, a = i - k * A_;
    wt[(size_t)a * R_ + k] = w[i];
  }
}

// ---------------------------------------------------------------------------
// bf16 WMMA GEMM, packed B, compile-time K/N so offsets fold into IOFFSET.
// Each wave: 1 M-tile x 4 N-tiles (A fragment reused 4x, 4 acc chains).
// unroll 2: bounded register pressure, still overlaps loads with WMMA.
// ---------------------------------------------------------------------------
template <int K, int N>
__global__ void gemm4_bf16_kernel(const __bf16* __restrict__ A,
                                  const __bf16* __restrict__ Bp,  // packed
                                  const float* __restrict__ bias,
                                  __bf16* __restrict__ outB,
                                  int M, int relu) {
  constexpr int NTN  = N >> 4;       // n-tiles
  constexpr int BSTR = NTN << 9;     // packed elements per k-slab
  constexpr int NJ   = NTN >> 2;     // n-jobs (groups of 4 tiles)
  int lane = threadIdx.x & 31;
  int wave = threadIdx.x >> 5;
  int job  = blockIdx.x * 8 + wave;
  int mt   = job / NJ;
  int ng   = job - mt * NJ;
  if (mt * 16 >= M) return;          // wave-uniform
  int m0 = mt * 16;
  int n0 = ng * 64;

  const __bf16* pa = A + (size_t)(m0 + (lane & 15)) * K + ((lane < 16) ? 0 : 8);
  const __bf16* pb = Bp + ((size_t)(ng * 4) << 9) + (lane << 4);

  v8f acc[4] = {};
#pragma unroll 2
  for (int ks = 0; ks < K / 32; ++ks) {
    v16bf a = make_a(pa + ks * 32);
    const __bf16* pbk = pb + (size_t)ks * BSTR;
#pragma unroll
    for (int u = 0; u < 4; ++u)
      acc[u] = wmma_bf16(a, *(const v16bf*)(pbk + u * 512), acc[u]);
  }

  int mh = (lane & 16) ? 8 : 0;
#pragma unroll
  for (int u = 0; u < 4; ++u) {
    int   n  = n0 + u * 16 + (lane & 15);
    float bv = bias[n];
#pragma unroll
    for (int i = 0; i < 8; ++i) {
      float v = acc[u][i] + bv;
      if (relu) v = fmaxf(v, 0.f);
      outB[(size_t)(m0 + mh + i) * N + n] = f2bf(v);
    }
  }
}

// ---------------------------------------------------------------------------
// Fused GRU step (one timestep). Grid = 8 blocks x 256 thr; each wave owns two
// 16(batch)x16(cols) tiles; 6 WMMA chains per tile (r/z/n x input/hidden).
// Only 64 waves in flight -> occupancy irrelevant: launch_bounds(256,1) gives
// the compiler the full VGPR budget (no spills); unroll capped at 2.
// sprev comes in already masked by done[t]; snext written pre-masked by
// done[t+1]. All strides compile-time constants.
// ---------------------------------------------------------------------------
__global__ void __launch_bounds__(256, 1)
gru_step_kernel(const __bf16* __restrict__ hid_t,  // [B,H]
                const __bf16* __restrict__ sprev,  // [B,R] masked
                __bf16* __restrict__ snext,        // [B,R]
                const float* __restrict__ hpf,     // [B,R] f32 raw
                const int* __restrict__ done_t,    // [B] this step
                const int* __restrict__ done_n,    // [B] next step
                const __bf16* __restrict__ wihp,   // packed [H,3R]
                const __bf16* __restrict__ whhp,   // packed [R,3R]
                const float* __restrict__ b_ih,    // [3R]
                const float* __restrict__ b_hh,    // [3R]
                float* __restrict__ out_t) {       // [B,R]
  constexpr int NTN  = (3 * R_) >> 4;   // 96 n-tiles
  constexpr int BSTR = NTN << 9;        // packed elements per k-slab
  constexpr int GOFF = 32 << 9;         // gate offset (32 n-tiles) in elements
  int lane = threadIdx.x & 31;
  int wave = threadIdx.x >> 5;
  int cblk = blockIdx.x * 64;

  for (int tt = wave; tt < 16; tt += 8) {   // wave-uniform
    int m0 = (tt >> 2) * 16;                // batch rows
    int c0 = cblk + (tt & 3) * 16;          // hidden cols
    int nt = c0 >> 4;

    const __bf16* pa  = hid_t + (size_t)(m0 + (lane & 15)) * H_ + ((lane < 16) ? 0 : 8);
    const __bf16* ps  = sprev + (size_t)(m0 + (lane & 15)) * R_ + ((lane < 16) ? 0 : 8);
    const __bf16* pwi = wihp + ((size_t)nt << 9) + (lane << 4);
    const __bf16* pwh = whhp + ((size_t)nt << 9) + (lane << 4);

    v8f air = {}, aiz = {}, ain = {};
    v8f ahr = {}, ahz = {}, ahn = {};
#pragma unroll 2
    for (int ks = 0; ks < H_ / 32; ++ks) {
      v16bf ax = make_a(pa + ks * 32);
      v16bf ah = make_a(ps + ks * 32);
      const __bf16* bi = pwi + (size_t)ks * BSTR;
      const __bf16* bh = pwh + (size_t)ks * BSTR;
      air = wmma_bf16(ax, *(const v16bf*)(bi),            air);
      aiz = wmma_bf16(ax, *(const v16bf*)(bi + GOFF),     aiz);
      ain = wmma_bf16(ax, *(const v16bf*)(bi + 2 * GOFF), ain);
      ahr = wmma_bf16(ah, *(const v16bf*)(bh),            ahr);
      ahz = wmma_bf16(ah, *(const v16bf*)(bh + GOFF),     ahz);
      ahn = wmma_bf16(ah, *(const v16bf*)(bh + 2 * GOFF), ahn);
    }

    int   c   = c0 + (lane & 15);
    float bir = b_ih[c], biz = b_ih[c + R_], bin = b_ih[c + 2 * R_];
    float bhr = b_hh[c], bhz = b_hh[c + R_], bhn = b_hh[c + 2 * R_];
    int   mh  = (lane & 16) ? 8 : 0;
#pragma unroll
    for (int i = 0; i < 8; ++i) {
      int    m   = m0 + mh + i;
      float  hp  = done_t[m] ? 0.f : hpf[(size_t)m * R_ + c];
      float  r   = sigmoidf_((air[i] + bir) + (ahr[i] + bhr));
      float  z   = sigmoidf_((aiz[i] + biz) + (ahz[i] + bhz));
      float  nn  = tanhf((ain[i] + bin) + r * (ahn[i] + bhn));
      float  hnw = (1.f - z) * nn + z * hp;
      size_t idx = (size_t)m * R_ + c;
      out_t[idx] = hnw;
      snext[idx] = done_n[m] ? (__bf16)0.f : f2bf(hnw);  // pre-mask for t+1
    }
  }
}

// ---------------------------------------------------------------------------
// Fused LayerNorm + actor (log-softmax, logprob gather, entropy) + critic.
// One wave per row; wave32 shuffle reductions; actor weights pre-transposed.
// ---------------------------------------------------------------------------
__global__ void heads_kernel(const float* __restrict__ hs,        // [TB,R]
                             const float* __restrict__ ln_g,
                             const float* __restrict__ ln_b,
                             const float* __restrict__ actor_wt,  // [A,R]
                             const float* __restrict__ actor_b,   // [A]
                             const float* __restrict__ critic_w,  // [R]
                             const float* __restrict__ critic_b,  // [1]
                             const int* __restrict__ action,      // [TB]
                             float* __restrict__ logprob,
                             float* __restrict__ entropy,
                             float* __restrict__ value) {
  int lane = threadIdx.x & 31;
  int wave = threadIdx.x >> 5;
  int row  = blockIdx.x * 8 + wave;
  const float* x = hs + (size_t)row * R_;

  float v[16];
  float s = 0.f;
#pragma unroll
  for (int j = 0; j < 16; ++j) { v[j] = x[lane + 32 * j]; s += v[j]; }
#pragma unroll
  for (int o = 16; o >= 1; o >>= 1) s += __shfl_xor(s, o, 32);
  float mu = s * (1.f / R_);

  float q = 0.f;
#pragma unroll
  for (int j = 0; j < 16; ++j) { float d = v[j] - mu; q += d * d; }
#pragma unroll
  for (int o = 16; o >= 1; o >>= 1) q += __shfl_xor(q, o, 32);
  float rstd = rsqrtf(q * (1.f / R_) + 1e-5f);

  float y[16];
#pragma unroll
  for (int j = 0; j < 16; ++j) {
    int k = lane + 32 * j;
    y[j] = (v[j] - mu) * rstd * ln_g[k] + ln_b[k];
  }

  float logits[A_];
#pragma unroll
  for (int a = 0; a < A_; ++a) {
    const float* wa = actor_wt + (size_t)a * R_;
    float t = 0.f;
#pragma unroll
    for (int j = 0; j < 16; ++j) t += y[j] * wa[lane + 32 * j];
#pragma unroll
    for (int o = 16; o >= 1; o >>= 1) t += __shfl_xor(t, o, 32);
    logits[a] = t + actor_b[a];
  }

  float mx = logits[0];
#pragma unroll
  for (int a = 1; a < A_; ++a) mx = fmaxf(mx, logits[a]);
  float se = 0.f;
#pragma unroll
  for (int a = 0; a < A_; ++a) se += expf(logits[a] - mx);
  float lse = mx + logf(se);

  int   act = action[row];
  float lp  = logits[act] - lse;
  float ent = 0.f;
#pragma unroll
  for (int a = 0; a < A_; ++a) {
    float la = logits[a] - lse;
    ent -= expf(la) * la;
  }

  float cv = 0.f;
#pragma unroll
  for (int j = 0; j < 16; ++j) cv += y[j] * critic_w[lane + 32 * j];
#pragma unroll
  for (int o = 16; o >= 1; o >>= 1) cv += __shfl_xor(cv, o, 32);
  cv += critic_b[0];

  if (lane == 0) {
    logprob[row] = lp;
    entropy[row] = ent;
    value[row]   = cv;
  }
}

__global__ void copy_f32_kernel(const float* __restrict__ in,
                                float* __restrict__ out, int n) {
  int i = blockIdx.x * blockDim.x + threadIdx.x;
  if (i < n) out[i] = in[i];
}

// ---------------------------------------------------------------------------
// host
// ---------------------------------------------------------------------------
extern "C" void kernel_launch(void* const* d_in, const int* in_sizes, int n_in,
                              void* d_out, int out_size, void* d_ws, size_t ws_size,
                              hipStream_t stream) {
  const float* x        = (const float*)d_in[0];
  const float* rnn0     = (const float*)d_in[1];
  const int*   done     = (const int*)d_in[2];
  const int*   action   = (const int*)d_in[3];
  const float* w1       = (const float*)d_in[4];
  const float* b1       = (const float*)d_in[5];
  const float* w2       = (const float*)d_in[6];
  const float* b2       = (const float*)d_in[7];
  const float* w3       = (const float*)d_in[8];
  const float* b3       = (const float*)d_in[9];
  const float* w_ih     = (const float*)d_in[10];
  const float* w_hh     = (const float*)d_in[11];
  const float* b_ih     = (const float*)d_in[12];
  const float* b_hh     = (const float*)d_in[13];
  const float* ln_g     = (const float*)d_in[14];
  const float* ln_b     = (const float*)d_in[15];
  const float* actor_w  = (const float*)d_in[16];
  const float* actor_b  = (const float*)d_in[17];
  const float* critic_w = (const float*)d_in[18];
  const float* critic_b = (const float*)d_in[19];

  float* out = (float*)d_out;
  float* o_logprob = out;
  float* o_entropy = out + TB_;
  float* o_value   = out + 2 * TB_;
  float* o_hfinal  = out + 3 * TB_;

  // workspace carve-up
  char*  ws  = (char*)d_ws;
  size_t off = 0;
  auto alloc = [&](size_t bytes) -> char* {
    char* p = ws + off;
    off = (off + bytes + 255) & ~(size_t)255;
    return p;
  };
  __bf16* wb1p  = (__bf16*)alloc((size_t)D_ * H_ * 2);          // packed
  __bf16* wb2p  = (__bf16*)alloc((size_t)H_ * (H_ / 2) * 2);    // packed
  __bf16* wb3p  = (__bf16*)alloc((size_t)(H_ / 2) * H_ * 2);    // packed
  __bf16* wihp  = (__bf16*)alloc((size_t)H_ * 3 * R_ * 2);      // packed
  __bf16* whhp  = (__bf16*)alloc((size_t)R_ * 3 * R_ * 2);      // packed
  __bf16* xb    = (__bf16*)alloc((size_t)TB_ * D_ * 2);
  __bf16* h1b   = (__bf16*)alloc((size_t)TB_ * H_ * 2);
  __bf16* h2b   = (__bf16*)alloc((size_t)TB_ * (H_ / 2) * 2);
  __bf16* hidb  = (__bf16*)alloc((size_t)TB_ * H_ * 2);
  float*  outf  = (float*)alloc((size_t)TB_ * R_ * 4);
  __bf16* sb0   = (__bf16*)alloc((size_t)B_ * R_ * 2);
  __bf16* sb1   = (__bf16*)alloc((size_t)B_ * R_ * 2);
  float*  awt   = (float*)alloc((size_t)A_ * R_ * 4);
  (void)ws_size; (void)n_in; (void)in_sizes; (void)out_size;

  auto pack = [&](const float* src, __bf16* dst, int K, int N) {
    int n = K * N;
    pack_b_kernel<<<(n + 255) / 256, 256, 0, stream>>>(src, dst, K, N);
  };
  pack(w1, wb1p, D_, H_);
  pack(w2, wb2p, H_, H_ / 2);
  pack(w3, wb3p, H_ / 2, H_);
  pack(w_ih, wihp, H_, 3 * R_);
  pack(w_hh, whhp, R_, 3 * R_);

  cvt_bf16_kernel<<<(TB_ * D_ + 255) / 256, 256, 0, stream>>>(x, xb, TB_ * D_);
  // initial bf16 state pre-masked by done[0]
  cvt_bf16_mask_kernel<<<(B_ * R_ + 255) / 256, 256, 0, stream>>>(
      rnn0, sb0, done, B_ * R_, R_);
  transpose_aw_kernel<<<(R_ * A_ + 255) / 256, 256, 0, stream>>>(actor_w, awt);

  // encoder MLP: three WMMA GEMMs (1 M-tile x 4 N-tiles per wave)
  gemm4_bf16_kernel<D_, H_><<<(TB_ / 16) * (H_ / 64) / 8, 256, 0, stream>>>(
      xb, wb1p, b1, h1b, TB_, 1);
  gemm4_bf16_kernel<H_, H_ / 2><<<(TB_ / 16) * ((H_ / 2) / 64) / 8, 256, 0, stream>>>(
      h1b, wb2p, b2, h2b, TB_, 1);
  gemm4_bf16_kernel<H_ / 2, H_><<<(TB_ / 16) * (H_ / 64) / 8, 256, 0, stream>>>(
      h2b, wb3p, b3, hidb, TB_, 1);

  // sequential GRU: one fused WMMA kernel per timestep (graph-replayed)
  __bf16* sbuf[2] = {sb0, sb1};
  for (int t = 0; t < T_; ++t) {
    const __bf16* hid_t = hidb + (size_t)t * B_ * H_;
    const __bf16* sprev = sbuf[t & 1];
    __bf16*       snext = sbuf[(t + 1) & 1];
    const float*  hpf   = (t == 0) ? rnn0 : (outf + (size_t)(t - 1) * B_ * R_);
    const int*    dn    = done + t * B_;
    const int*    dnn   = done + ((t + 1 < T_) ? (t + 1) : t) * B_;
    float*        ot    = outf + (size_t)t * B_ * R_;
    gru_step_kernel<<<R_ / 64, 256, 0, stream>>>(hid_t, sprev, snext, hpf, dn,
                                                 dnn, wihp, whhp, b_ih, b_hh, ot);
  }

  // LayerNorm + heads (one wave per row)
  heads_kernel<<<TB_ / 8, 256, 0, stream>>>(outf, ln_g, ln_b, awt, actor_b,
                                            critic_w, critic_b, action,
                                            o_logprob, o_entropy, o_value);

  // h_final = hidden state after last step
  copy_f32_kernel<<<(B_ * R_ + 255) / 256, 256, 0, stream>>>(
      outf + (size_t)(T_ - 1) * B_ * R_, o_hfinal, B_ * R_);
}
